// Rnn_21741124452621
// MI455X (gfx1250) — matrix operand order repaired and checked by reading, not compiled, not run on hardware
//
#include <hip/hip_runtime.h>

typedef __attribute__((ext_vector_type(16))) _Float16 v16h;
typedef __attribute__((ext_vector_type(8)))  float    v8f;
typedef __attribute__((ext_vector_type(8)))  unsigned int v8u;
typedef __attribute__((ext_vector_type(4)))  float    f4;

#define LSTM_B 4096
#define LSTM_T 512
#define LSTM_I 11
#define LSTM_H 16

#define LOG2E  1.4426950408889634f

// Prefer the CDNA5 hardware tanh (V_TANH_F32, single TRANS op) when the
// toolchain exposes it; otherwise fall back to exp2+rcp (2 TRANS ops).
#if __has_builtin(__builtin_amdgcn_tanhf)
__device__ __forceinline__ float tanhfast(float x) {
    return __builtin_amdgcn_tanhf(x);
}
#else
__device__ __forceinline__ float tanhfast(float x) {
    return 2.0f * __builtin_amdgcn_rcpf(1.0f + __builtin_amdgcn_exp2f(-2.0f * LOG2E * x)) - 1.0f;
}
#endif

// sigmoid(x) = 0.5*tanh(x/2) + 0.5  (mul + tanh + fma on the hw-tanh path)
__device__ __forceinline__ float sigf(float x) {
    return fmaf(0.5f, tanhfast(0.5f * x), 0.5f);
}

__device__ __forceinline__ v8f wmma_f16(v16h a, v16h b, v8f c) {
    // D = A(16x32 f16) * B(32x16 f16) + C(16x16 f32)
    return __builtin_amdgcn_wmma_f32_16x16x32_f16(false, a, false, b,
                                                  (short)0, c, false, false);
}

// pack 8 f32 lane-half values (v_cvt_pk_rtz_f16_f32), exchange with the partner lane
// (xor 16), and build the full 16-feature f16 vector of this lane's batch column.
__device__ __forceinline__ v16h exchange_pack(const float hv[8], bool hi) {
    unsigned int p[4], q[4];
#pragma unroll
    for (int r = 0; r < 4; ++r) {
        auto ph = __builtin_amdgcn_cvt_pkrtz(hv[2 * r], hv[2 * r + 1]);
        p[r] = __builtin_bit_cast(unsigned int, ph);
        q[r] = __shfl_xor(p[r], 16, 32);
    }
    v8u u;
#pragma unroll
    for (int r = 0; r < 4; ++r) {
        u[r]     = hi ? q[r] : p[r];   // features 0..7
        u[r + 4] = hi ? p[r] : q[r];   // features 8..15
    }
    return __builtin_bit_cast(v16h, u);
}

__global__ void __launch_bounds__(128)
lstm2_softmax_kernel(const float* __restrict__ x,
                     const float* __restrict__ w_ih0, const float* __restrict__ w_hh0,
                     const float* __restrict__ b_ih0, const float* __restrict__ b_hh0,
                     const float* __restrict__ w_ih1, const float* __restrict__ w_hh1,
                     const float* __restrict__ b_ih1, const float* __restrict__ b_hh1,
                     float* __restrict__ out)
{
    const int lane = threadIdx.x & 31;
    const int wave = blockIdx.x * (blockDim.x >> 5) + (threadIdx.x >> 5);
    const int n    = lane & 15;            // batch column within the 16-row tile
    const int b    = wave * 16 + n;        // global batch row for this lane
    const bool hi  = (lane >= 16);
    const int mOff = hi ? 8 : 0;           // D-fragment row offset for this lane half

    // ---------------- loop-invariant A fragments (weights) + bias C fragments ----
    // A tile g covers gate rows [16g, 16g+16) with K = [W_ih (0..15, I padded) | W_hh (16..31)].
    v16h A0[4], A1[4];
    v8f  CB0[4], CB1[4];
    const int M = lane & 15;
#pragma unroll
    for (int g = 0; g < 4; ++g) {
        const int row = g * 16 + M;
        v16h a0 = {}, a1 = {};
#pragma unroll
        for (int e = 0; e < 16; ++e) {
            const int k = e + (hi ? 16 : 0);   // 0..31
            float w0, w1;
            if (k < 16) {   // input-weight half (layer0 pads K=11..15 with zero)
                w0 = (k < LSTM_I) ? w_ih0[row * LSTM_I + k] : 0.0f;
                w1 = w_ih1[row * LSTM_H + k];
            } else {        // recurrent-weight half
                w0 = w_hh0[row * LSTM_H + (k - 16)];
                w1 = w_hh1[row * LSTM_H + (k - 16)];
            }
            a0[e] = (_Float16)w0;
            a1[e] = (_Float16)w1;
        }
        A0[g] = a0; A1[g] = a1;
        v8f cb0, cb1;
#pragma unroll
        for (int r = 0; r < 8; ++r) {
            const int m = g * 16 + r + mOff;
            cb0[r] = b_ih0[m] + b_hh0[m];
            cb1[r] = b_ih1[m] + b_hh1[m];
        }
        CB0[g] = cb0; CB1[g] = cb1;
    }

    // ---------------- recurrent state (all in VGPRs) ----------------------------
    v16h h0f = {};          // full 16-feature f16 vector of h0 for this lane's column
    v16h h1f = {};          // same for h1
    v8f  c0  = {};          // cell state fragments (f32, D layout)
    v8f  c1  = {};
    float hv1[8] = {};      // last layer-1 h values (f32) for the final softmax

    const float* xrow = x + (size_t)b * LSTM_T * LSTM_I;

    for (int t = 0; t < LSTM_T; ++t) {
        // ---- x_t for this column: lanes 0-15 supply the input half of B0 -------
        v16h xf = {};
        if (!hi) {
            const float* xp = xrow + (size_t)t * LSTM_I;
            if (t + 8 < LSTM_T)
                __builtin_prefetch(xp + 8 * LSTM_I, 0, 1);   // global_prefetch_b8
#pragma unroll
            for (int e = 0; e < LSTM_I; ++e) xf[e] = (_Float16)xp[e];
        }

        // ================= layer 0:  gates = [Wih0|Whh0] x [x_t ; h0_{t-1}] + b =
        const v16h B0v = hi ? h0f : xf;
        v8f di = wmma_f16(A0[0], B0v, CB0[0]);
        v8f df = wmma_f16(A0[1], B0v, CB0[1]);
        v8f dg = wmma_f16(A0[2], B0v, CB0[2]);
        v8f dw = wmma_f16(A0[3], B0v, CB0[3]);

        float hv0[8];
#pragma unroll
        for (int r = 0; r < 8; ++r) {
            const float ig = sigf(di[r]);
            const float fg = sigf(df[r]);
            const float gg = tanhfast(dg[r]);
            const float og = sigf(dw[r]);
            const float cn = fg * c0[r] + ig * gg;
            c0[r]  = cn;
            hv0[r] = og * tanhfast(cn);
        }
        h0f = exchange_pack(hv0, hi);

        // ================= layer 1:  gates = [Wih1|Whh1] x [h0_t ; h1_{t-1}] + b =
        const v16h B1v = hi ? h1f : h0f;
        v8f ei = wmma_f16(A1[0], B1v, CB1[0]);
        v8f ef = wmma_f16(A1[1], B1v, CB1[1]);
        v8f eg = wmma_f16(A1[2], B1v, CB1[2]);
        v8f ew = wmma_f16(A1[3], B1v, CB1[3]);

#pragma unroll
        for (int r = 0; r < 8; ++r) {
            const float ig = sigf(ei[r]);
            const float fg = sigf(ef[r]);
            const float gg = tanhfast(eg[r]);
            const float og = sigf(ew[r]);
            const float cn = fg * c1[r] + ig * gg;
            c1[r]  = cn;
            hv1[r] = og * tanhfast(cn);
        }
        h1f = exchange_pack(hv1, hi);
    }

    // ---------------- softmax over the 16 features of h1[:, T-1] ----------------
    float mx = hv1[0];
#pragma unroll
    for (int r = 1; r < 8; ++r) mx = fmaxf(mx, hv1[r]);
    mx = fmaxf(mx, __shfl_xor(mx, 16, 32));       // combine lane halves

    float ex[8];
    float s = 0.0f;
#pragma unroll
    for (int r = 0; r < 8; ++r) {
        ex[r] = __builtin_amdgcn_exp2f(LOG2E * (hv1[r] - mx));
        s += ex[r];
    }
    s += __shfl_xor(s, 16, 32);
    const float inv = __builtin_amdgcn_rcpf(s);

    // two 16B stores (32B-aligned address)
    f4 o0 = { ex[0] * inv, ex[1] * inv, ex[2] * inv, ex[3] * inv };
    f4 o1 = { ex[4] * inv, ex[5] * inv, ex[6] * inv, ex[7] * inv };
    f4* op = (f4*)(out + (size_t)b * LSTM_H + mOff);
    op[0] = o0;
    op[1] = o1;
}

extern "C" void kernel_launch(void* const* d_in, const int* in_sizes, int n_in,
                              void* d_out, int out_size, void* d_ws, size_t ws_size,
                              hipStream_t stream) {
    const float* x     = (const float*)d_in[0];
    const float* w_ih0 = (const float*)d_in[1];
    const float* w_hh0 = (const float*)d_in[2];
    const float* b_ih0 = (const float*)d_in[3];
    const float* b_hh0 = (const float*)d_in[4];
    const float* w_ih1 = (const float*)d_in[5];
    const float* w_hh1 = (const float*)d_in[6];
    const float* b_ih1 = (const float*)d_in[7];
    const float* b_hh1 = (const float*)d_in[8];
    float* out = (float*)d_out;

    // 16 batch rows per wave, 4 waves per block -> 64 rows/block, 64 blocks
    dim3 block(128);
    dim3 grid(LSTM_B / 64);
    lstm2_softmax_kernel<<<grid, block, 0, stream>>>(
        x, w_ih0, w_hh0, b_ih0, b_hh0, w_ih1, w_hh1, b_ih1, b_hh1, out);
}
